// Model_77962246357547
// MI455X (gfx1250) — matrix operand order repaired and checked by reading, not compiled, not run on hardware
//
#include <hip/hip_runtime.h>
#include <hip/hip_bf16.h>

typedef __attribute__((ext_vector_type(16))) __bf16 v16bf;
typedef __attribute__((ext_vector_type(8)))  __bf16 v8bf;
typedef __attribute__((ext_vector_type(8)))  float  v8f;

#define TT    2048
#define BB    2048
#define FIN   10      // M+1 input features
#define KH    128     // hidden size
#define MOUT  9       // output features
#define KPAD  160     // 128 (h) + 32 (x padded) -> 5 K-tiles of 32
#define NT_G  32      // 512 gate cols / 16
#define NT    33      // + 1 output-projection tile
#define BTILE 16      // batch rows per workgroup
#define NTHREADS 256  // 8 wave32 waves

static constexpr int SB_ELEMS = NT * 5 * 32 * 16;  // 84480 bf16 (staging, stays resident)
static constexpr int SA_ELEMS = BTILE * KPAD;      // 2560 bf16 per buffer

// Fast activations: v_exp_f32 + v_rcp_f32 (TRANS pipe, co-executes with XDL WMMA),
// avoiding the precise-division expansion (div_scale/rcp/fma/fmas/fixup).
__device__ __forceinline__ float sigmoidf_(float x) {
  return __builtin_amdgcn_rcpf(1.0f + __expf(-x));
}
__device__ __forceinline__ float tanhf_(float x) {
  return 2.0f * __builtin_amdgcn_rcpf(1.0f + __expf(-2.0f * x)) - 1.0f;
}

// A fragment: 16x32 bf16; lane(0-15)=row M with K-base 0, lanes 16-31 K-base 8;
// two contiguous 16B runs per lane -> 2x ds_load_b128.
__device__ __forceinline__ v16bf load_a_frag(const __bf16* sA, int lane, int kt) {
  const int row = lane & 15;
  const int kb  = (lane >> 4) << 3;
  union { v16bf v; v8bf h[2]; } u;
  const __bf16* p = sA + row * KPAD + kt * 32 + kb;
  u.h[0] = *(const v8bf*)(p);
  u.h[1] = *(const v8bf*)(p + 16);
  return u.v;
}

__device__ __forceinline__ v16bf load_b_frag(const __bf16* sB, int n, int kt, int lane) {
  return *(const v16bf*)(sB + (((n * 5 + kt) * 32 + lane) << 4));
}

__global__ __launch_bounds__(NTHREADS, 1) __attribute__((amdgpu_waves_per_eu(2)))
void lstm_fused_kernel(const float* __restrict__ inputs,
                       const float* __restrict__ W_ii, const float* __restrict__ W_hi,
                       const float* __restrict__ W_if, const float* __restrict__ W_hf,
                       const float* __restrict__ W_ig, const float* __restrict__ W_hg,
                       const float* __restrict__ W_io, const float* __restrict__ W_ho,
                       const float* __restrict__ V_w,  const float* __restrict__ V_b,
                       float* __restrict__ out) {
  extern __shared__ __align__(16) char smem_raw[];
  __bf16* sB  = (__bf16*)smem_raw;        // weight fragments (B-layout), resident
  __bf16* sA0 = sB + SB_ELEMS;            // A tile double buffer
  __bf16* sA1 = sA0 + SA_ELEMS;

  const int tid   = threadIdx.x;
  const int lane  = tid & 31;
  const int wavei = __builtin_amdgcn_readfirstlane(tid >> 5);  // SGPR -> scalar branches
  const int b0    = blockIdx.x * BTILE;

  // ---- one-time: stage combined weights [Wh;Wx;0 | Vw] as bf16 B-fragments ----
  const float* Whs[4] = {W_hi, W_hf, W_hg, W_ho};
  const float* Wxs[4] = {W_ii, W_if, W_ig, W_io};
  for (int e = tid; e < SB_ELEMS; e += NTHREADS) {
    int n    = e / (5 * 32 * 16);
    int r1   = e % (5 * 32 * 16);
    int kt   = r1 / (32 * 16);
    int r2   = r1 % (32 * 16);
    int l    = r2 >> 4;
    int i    = r2 & 15;
    int col  = l & 15;
    int krow = kt * 32 + ((l >> 4) << 3) + (i & 7) + ((i >> 3) << 4);
    float v = 0.0f;
    if (n < NT_G) {
      int gc = n * 16 + col;
      int q  = gc >> 7;
      int kc = gc & 127;
      if (krow < KH)            v = Whs[q][krow * KH + kc];
      else if (krow < KH + FIN) v = Wxs[q][(krow - KH) * KH + kc];
    } else if (krow < KH && col < MOUT) {
      v = V_w[krow * MOUT + col];
    }
    sB[e] = (__bf16)v;
  }
  for (int e = tid; e < 2 * SA_ELEMS; e += NTHREADS) sA0[e] = (__bf16)0.0f;

  // x software pipeline: thread tid<160 owns (row, f)
  const int xrow = tid / FIN;
  const int xf   = tid % FIN;
  float x0 = 0.0f;
  if (tid < BTILE * FIN)
    x0 = __builtin_nontemporal_load(&inputs[(size_t)(b0 + xrow) * (TT * FIN) + xf]);

  __syncthreads();  // sB staged

  // ---- hoist the wave's time-invariant gate weights into VGPRs (20 frags = 160 VGPRs) ----
  v16bf wreg[20];
  #pragma unroll
  for (int q = 0; q < 4; ++q)
    #pragma unroll
    for (int kt = 0; kt < 5; ++kt)
      wreg[q * 5 + kt] = load_b_frag(sB, q * 8 + wavei, kt, lane);

  // seed buffer 0 with x_0; prefetch x_1
  if (tid < BTILE * FIN) sA0[xrow * KPAD + KH + xf] = (__bf16)x0;
  float xr = 0.0f;
  if (tid < BTILE * FIN)
    xr = __builtin_nontemporal_load(&inputs[(size_t)(b0 + xrow) * (TT * FIN) + 1 * FIN + xf]);

  const int   mcol   = lane & 15;
  const int   rowoff = (lane >> 4) << 3;
  const float vb     = (mcol < MOUT) ? V_b[mcol] : 0.0f;

  float creg[8], hreg[8];
  #pragma unroll
  for (int r = 0; r < 8; ++r) { creg[r] = 0.0f; hreg[r] = 0.0f; }

  __syncthreads();  // buffer 0 ready

  for (int t = 0; t <= TT; ++t) {
    __bf16* sAc = (t & 1) ? sA1 : sA0;
    __bf16* sAn = (t & 1) ? sA0 : sA1;

    // stage x_{t+1} (loaded a full step ago) into next buffer; issue load of x_{t+2}
    if (t + 1 < TT && tid < BTILE * FIN) sAn[xrow * KPAD + KH + xf] = (__bf16)xr;
    if (t + 2 < TT && tid < BTILE * FIN)
      xr = __builtin_nontemporal_load(
          &inputs[(size_t)(b0 + xrow) * (TT * FIN) + (size_t)(t + 2) * FIN + xf]);

    v8f oacc = {};
    if (t < TT) {
      v8f gacc[4] = {};
      #pragma unroll
      for (int kt = 0; kt < 5; ++kt) {
        v16bf a = load_a_frag(sAc, lane, kt);
        #pragma unroll
        for (int q = 0; q < 4; ++q)
          gacc[q] = __builtin_amdgcn_wmma_f32_16x16x32_bf16(
              false, a, false, wreg[q * 5 + kt], (short)0, gacc[q], false, false);
        if (wavei == 7 && kt < 4) {  // out[t-1] = h_{t-1} @ Vw (padded rows are zero)
          v16bf bo = load_b_frag(sB, 32, kt, lane);
          oacc = __builtin_amdgcn_wmma_f32_16x16x32_bf16(
              false, a, false, bo, (short)0, oacc, false, false);
        }
      }

      // pointwise LSTM cell update (fp32, wave-local 16 gate columns)
      #pragma unroll
      for (int r = 0; r < 8; ++r) {
        float iv = sigmoidf_(gacc[0][r]);
        float fv = sigmoidf_(gacc[1][r]);
        float gv = tanhf_(gacc[2][r]);
        float ov = sigmoidf_(gacc[3][r]);
        float cn = fv * creg[r] + iv * gv;
        creg[r] = cn;
        hreg[r] = ov * tanhf_(cn);
      }
      // publish h_t into next A buffer (wave w owns cols [w*16, w*16+16))
      #pragma unroll
      for (int r = 0; r < 8; ++r)
        sAn[(r + rowoff) * KPAD + wavei * 16 + mcol] = (__bf16)hreg[r];
    } else if (wavei == 7) {  // epilogue: out[T-1]
      #pragma unroll
      for (int kt = 0; kt < 4; ++kt) {
        v16bf a  = load_a_frag(sAc, lane, kt);
        v16bf bo = load_b_frag(sB, 32, kt, lane);
        oacc = __builtin_amdgcn_wmma_f32_16x16x32_bf16(
            false, a, false, bo, (short)0, oacc, false, false);
      }
    }

    if (wavei == 7 && t > 0 && mcol < MOUT) {
      #pragma unroll
      for (int r = 0; r < 8; ++r)
        __builtin_nontemporal_store(
            oacc[r] + vb,
            &out[(size_t)(b0 + r + rowoff) * (TT * MOUT) + (size_t)(t - 1) * MOUT + mcol]);
    }

    __syncthreads();  // single barrier per step (double-buffered A tile)
  }
}

extern "C" void kernel_launch(void* const* d_in, const int* in_sizes, int n_in,
                              void* d_out, int out_size, void* d_ws, size_t ws_size,
                              hipStream_t stream) {
  const float* inputs = (const float*)d_in[0];
  const float* W_ii   = (const float*)d_in[1];
  const float* W_hi   = (const float*)d_in[2];
  const float* W_if   = (const float*)d_in[3];
  const float* W_hf   = (const float*)d_in[4];
  const float* W_ig   = (const float*)d_in[5];
  const float* W_hg   = (const float*)d_in[6];
  const float* W_io   = (const float*)d_in[7];
  const float* W_ho   = (const float*)d_in[8];
  const float* V_w    = (const float*)d_in[9];
  const float* V_b    = (const float*)d_in[10];
  float* out = (float*)d_out;

  const size_t smem = (size_t)(SB_ELEMS + 2 * SA_ELEMS) * sizeof(__bf16);  // ~175 KB of 320 KB
  dim3 grid(BB / BTILE);     // 128 independent workgroups (batch-parallel recurrence)
  dim3 block(NTHREADS);      // 8 wave32 waves -> 2 waves/SIMD
  lstm_fused_kernel<<<grid, block, smem, stream>>>(
      inputs, W_ii, W_hi, W_if, W_hf, W_ig, W_hg, W_io, W_ho, V_w, V_b, out);
}
